// DMPNNModel_36816459662025
// MI455X (gfx1250) — matrix operand order repaired and checked by reading, not compiled, not run on hardware
//
#include <hip/hip_runtime.h>
#include <hip/hip_bf16.h>
#include <stdint.h>
#include <stddef.h>

// Problem sizes (fixed by the reference).
#define KN 4096      // nodes
#define KE 12288     // edges
#define KFN 64       // node feature dim
#define KFE 16       // edge feature dim
#define KH 128       // hidden
#define KL 4         // layers
#define KG 256       // graphs
#define BN_EPS 1e-5f

typedef __bf16 bf16;
typedef __attribute__((ext_vector_type(16))) __bf16 v16bf;
typedef __attribute__((ext_vector_type(8)))  float  v8f;
typedef unsigned int u32x4 __attribute__((ext_vector_type(4)));
typedef float f32x4 __attribute__((ext_vector_type(4)));
typedef int i32x4 __attribute__((ext_vector_type(4)));

union V16 { v16bf v; bf16 h[16]; u32x4 u[2]; };
union BF2 { bf16 b[2]; unsigned int u; };

#if __has_builtin(__builtin_amdgcn_global_load_async_to_lds_b128)
#define HAVE_ASYNC_LDS 1
#endif

static __device__ __forceinline__ void cp_async_b128(const void* g, void* l) {
#ifdef HAVE_ASYNC_LDS
  // GLOBAL_LOAD_ASYNC_TO_LDS_B128: per-lane 16B global->LDS, tracked by ASYNCcnt.
  __builtin_amdgcn_global_load_async_to_lds_b128((i32x4*)g, (i32x4*)l, 0, 0);
#else
  *(u32x4*)l = *(const u32x4*)g;
#endif
}

static __device__ __forceinline__ void wait_async0() {
#ifdef HAVE_ASYNC_LDS
#if __has_builtin(__builtin_amdgcn_s_wait_asynccnt)
  __builtin_amdgcn_s_wait_asynccnt(0);
#else
  asm volatile("s_wait_asynccnt 0" ::: "memory");
#endif
#endif
}

static __device__ __forceinline__ v8f wmma_bf16(v16bf a, v16bf b, v8f c) {
  // D = A(16x32 bf16) x B(32x16 bf16) + C(16x16 f32)
  return __builtin_amdgcn_wmma_f32_16x16x32_bf16(false, a, false, b, (short)0, c,
                                                 false, false);
}

static __device__ __forceinline__ v8f vzero8() {
  v8f z;
#pragma unroll
  for (int i = 0; i < 8; ++i) z[i] = 0.f;
  return z;
}

// ---------------------------------------------------------------------------
// Swizzle fp32 [K x 128] row-major weight -> bf16 WMMA-B fragment order:
//   dst[((c*8+nt)*32+lane)*16 + j]  where kappa = c*32 + (lane>>4)*8 + (j>>3)*16 + (j&7),
//   o = nt*16 + (lane&15).  Lanes 0-15 hold K 0..7,16..23; lanes 16-31 hold 8..15,24..31.
// Zero-pads kappa >= Kreal (used for K=16 edge MLP padded to 32).
// ---------------------------------------------------------------------------
__global__ __launch_bounds__(256) void k_swizzle_b(const float* __restrict__ src,
                                                   bf16* __restrict__ dst,
                                                   int Kc, int Kreal) {
  int t = blockIdx.x * 256 + threadIdx.x;       // one thread = 2 bf16 elements
  int total = Kc * 2048;
  if (t >= total) return;
  int j2   = t & 7;
  int lane = (t >> 3) & 31;
  int nt   = (t >> 8) & 7;
  int c    = t >> 11;
  int o    = nt * 16 + (lane & 15);
  int kap0 = c * 32 + (lane >> 4) * 8 + (j2 >> 2) * 16 + (j2 & 3) * 2;
  float f0 = (kap0     < Kreal) ? src[(size_t)kap0 * 128 + o]       : 0.f;
  float f1 = (kap0 + 1 < Kreal) ? src[(size_t)(kap0 + 1) * 128 + o] : 0.f;
  BF2 p; p.b[0] = (bf16)f0; p.b[1] = (bf16)f1;
  ((unsigned int*)dst)[t] = p.u;                // coalesced b32 stores
}

// ---------------------------------------------------------------------------
// Generic WMMA GEMM:  out[M x 128] = A[M x KREAL] (fp32) @ Bsw + bias
// KREAL/KC are compile-time so the A fragment is branch-free vector loads.
// Each wave computes a 16x128 tile; 8 waves / block; M multiple of 128.
// ---------------------------------------------------------------------------
template <int KREAL, int KC, int RELU>
__global__ __launch_bounds__(256) void k_gemm(const float* __restrict__ A,
                                              const bf16* __restrict__ Bsw,
                                              const float* __restrict__ bias,
                                              float* __restrict__ outF,
                                              bf16* __restrict__ outB) {
  int tid  = threadIdx.x;
  int wave = tid >> 5, lane = tid & 31;
  int m = lane & 15, half = lane >> 4;
  int row0 = (blockIdx.x * 8 + wave) * 16;
  const float* arow = A + (size_t)(row0 + m) * KREAL;

  v8f acc[8];
#pragma unroll
  for (int nt = 0; nt < 8; ++nt) acc[nt] = vzero8();

  const u32x4* bp4 = (const u32x4*)Bsw;
#pragma unroll
  for (int c = 0; c < KC; ++c) {
    // A fragment (16x32 bf16): lanes 0-15 j0..7 -> K c*32+half*8.., j8..15 -> +16.
    V16 af;
    {
      const f32x4* lo = (const f32x4*)(arow + c * 32 + half * 8);
      f32x4 l0 = lo[0], l1 = lo[1];
#pragma unroll
      for (int j = 0; j < 4; ++j) af.h[j]     = (bf16)l0[j];
#pragma unroll
      for (int j = 0; j < 4; ++j) af.h[4 + j] = (bf16)l1[j];
      if constexpr ((KREAL & 31) == 0) {
        const f32x4* hi = (const f32x4*)(arow + c * 32 + 16 + half * 8);
        f32x4 h0 = hi[0], h1 = hi[1];
#pragma unroll
        for (int j = 0; j < 4; ++j) af.h[8 + j]  = (bf16)h0[j];
#pragma unroll
        for (int j = 0; j < 4; ++j) af.h[12 + j] = (bf16)h1[j];
      } else {
        // KREAL==16 padded to one 32-chunk: j>=8 maps to k>=16 -> zero.
#pragma unroll
        for (int j = 8; j < 16; ++j) af.h[j] = (bf16)0.f;
      }
    }
    // Preload all 8 B fragments (clauses), then issue the 8 WMMAs.
    V16 bfr[8];
#pragma unroll
    for (int nt = 0; nt < 8; ++nt) {
      size_t q = (size_t)((c * 8 + nt) * 32 + lane) * 2;  // u32x4 units
      bfr[nt].u[0] = bp4[q];
      bfr[nt].u[1] = bp4[q + 1];
    }
#pragma unroll
    for (int nt = 0; nt < 8; ++nt)
      acc[nt] = wmma_bf16(af.v, bfr[nt].v, acc[nt]);
  }
#pragma unroll
  for (int nt = 0; nt < 8; ++nt) {
    int o = nt * 16 + m;
    float bo = bias ? bias[o] : 0.f;
#pragma unroll
    for (int r = 0; r < 8; ++r) {
      int row = row0 + half * 8 + r;
      float v = acc[nt][r] + bo;
      if (RELU) v = fmaxf(v, 0.f);
      size_t idx = (size_t)row * 128 + o;
      if (outF) outF[idx] = v;
      if (outB) outB[idx] = (bf16)v;
    }
  }
}

// ---------------------------------------------------------------------------
// Fused NNConv message GEMM + mean-aggregation scatter.
//   msg[e,o] = sum_{kappa=(k,h)} e[e,k]*h[src[e],h] * e2w[kappa,o]  (+ h_src@e2_b)
// A built in registers (rank-1: bf16 splat-multiply); B (e2_w bf16, swizzled,
// 4MB -> L2 resident) double-buffered through LDS via async global->LDS copies
// overlapping the WMMAs.  Block = 8 waves = 128 edges; each wave owns a
// 16-edge x 128-out tile.  kappa = k_e*128 + h; chunk c = k_e*4 + h0c so the
// per-wave h fragment is reused across all 128 k_e steps.
// ---------------------------------------------------------------------------
__global__ __launch_bounds__(256) void k_fused_msg(const bf16* __restrict__ hbf,
                                                   const bf16* __restrict__ ebf,
                                                   const bf16* __restrict__ Bsw,
                                                   const int* __restrict__ srcI,
                                                   const int* __restrict__ dstI,
                                                   const float* __restrict__ hb,
                                                   float* __restrict__ agg) {
  __shared__ bf16  eLds[128 * 128];   // 128 edges x 128 feats = 32 KB
  __shared__ u32x4 Bld[2][512];       // double-buffered 8 KB B chunks

  int tid  = threadIdx.x;
  int wave = tid >> 5, lane = tid & 31;
  int m = lane & 15, half = lane >> 4;
  int ebase = blockIdx.x * 128;

  // Stage this block's e rows (contiguous 32 KB) + first B chunk, async.
  {
    const u32x4* g = (const u32x4*)(ebf + (size_t)ebase * 128);
    u32x4* s = (u32x4*)eLds;
#pragma unroll
    for (int i = 0; i < 8; ++i) cp_async_b128(&g[tid + i * 256], &s[tid + i * 256]);
    const u32x4* gb = (const u32x4*)Bsw;
    cp_async_b128(&gb[tid],       &Bld[0][tid]);
    cp_async_b128(&gb[tid + 256], &Bld[0][tid + 256]);
  }
  int myedge = ebase + wave * 16 + m;
  int mynode = srcI[myedge];
  const u32x4* hrow = (const u32x4*)(hbf + (size_t)mynode * 128);

  v8f acc[8];
#pragma unroll
  for (int nt = 0; nt < 8; ++nt) acc[nt] = vzero8();

  wait_async0();
  __syncthreads();

  int t = 0;
  for (int h0c = 0; h0c < 4; ++h0c) {
    // h fragment for this 32-wide h chunk (A layout); convert to f32 once,
    // reused for 128 k_e steps.
    V16 hf;
    hf.u[0] = hrow[h0c * 4 + half];
    hf.u[1] = hrow[h0c * 4 + 2 + half];
    float hfl[16];
#pragma unroll
    for (int j = 0; j < 16; ++j) hfl[j] = (float)hf.h[j];
#pragma unroll 1
    for (int ke = 0; ke < 128; ++ke) {
      int cur = t & 1;
      if (t + 1 < 512) {                       // async-prefetch next chunk
        int tn = t + 1;
        int cn = (tn & 127) * 4 + (tn >> 7);
        const u32x4* g = (const u32x4*)(Bsw + (size_t)cn * 4096);
        cp_async_b128(&g[tid],       &Bld[cur ^ 1][tid]);
        cp_async_b128(&g[tid + 256], &Bld[cur ^ 1][tid + 256]);
      }
      // A = e_scalar * h_frag  (f32 mul, packed cvt to bf16).
      float ev = (float)eLds[(wave * 16 + m) * 128 + ke];
      V16 af;
#pragma unroll
      for (int j = 0; j < 16; ++j) af.h[j] = (bf16)(ev * hfl[j]);

      // Preload the 8 B fragments from LDS, then 8 WMMAs.
      const u32x4* bp4 = (const u32x4*)&Bld[cur][0];
      V16 bfr[8];
#pragma unroll
      for (int nt = 0; nt < 8; ++nt) {
        int q = (nt * 32 + lane) * 2;
        bfr[nt].u[0] = bp4[q];
        bfr[nt].u[1] = bp4[q + 1];
      }
#pragma unroll
      for (int nt = 0; nt < 8; ++nt)
        acc[nt] = wmma_bf16(af.v, bfr[nt].v, acc[nt]);

      wait_async0();            // own async copies landed in LDS
      __syncthreads();          // whole block's copies visible
      ++t;
    }
  }

  // Scatter: agg[dst] += msg (+ per-edge e2_b bias term h_src @ e2_b).
#pragma unroll
  for (int r = 0; r < 8; ++r) {
    int ed = ebase + wave * 16 + half * 8 + r;  // C layout: M = r + half*8
    int dn = dstI[ed];
    int sn = srcI[ed];
    const float* hbrow = hb + (size_t)sn * 128;
    float* arow = agg + (size_t)dn * 128;
#pragma unroll
    for (int nt = 0; nt < 8; ++nt) {
      int o = nt * 16 + m;
      atomicAdd(&arow[o], acc[nt][r] + hbrow[o]);
    }
  }
}

// --------------------------- small fp32 kernels ----------------------------
__global__ void k_fill(float* p, int n) {
  int i = blockIdx.x * 256 + threadIdx.x;
  if (i < n) p[i] = 0.f;
}
__global__ void k_deg(const int* __restrict__ dstI, float* __restrict__ deg) {
  int e = blockIdx.x * 256 + threadIdx.x;
  if (e < KE) atomicAdd(&deg[dstI[e]], 1.f);
}
__global__ void k_invdeg(const float* __restrict__ deg, float* __restrict__ inv) {
  int n = blockIdx.x * 256 + threadIdx.x;
  if (n < KN) { float d = deg[n]; inv[n] = d > 0.f ? 1.f / fmaxf(d, 1.f) : 0.f; }
}
// out = agg*inv_deg + root + conv_b; accumulate column sums for BN stats.
__global__ __launch_bounds__(256) void k_combine1(
    const float* __restrict__ agg, const float* __restrict__ invdeg,
    const float* __restrict__ rootout, const float* __restrict__ convb,
    float* __restrict__ out, float* __restrict__ colsum,
    float* __restrict__ colsumsq) {
  int tid = threadIdx.x;
  int o = tid & 127, rh = tid >> 7;
  int rbase = blockIdx.x * 32;
  float s = 0.f, q = 0.f;
  float cb = convb[o];
  for (int r = rbase + rh; r < rbase + 32; r += 2) {
    size_t idx = (size_t)r * 128 + o;
    float v = agg[idx] * invdeg[r] + rootout[idx] + cb;
    out[idx] = v; s += v; q += v * v;
  }
  __shared__ float ls[256], lq[256];
  ls[tid] = s; lq[tid] = q;
  __syncthreads();
  if (tid < 128) {
    atomicAdd(&colsum[o],   ls[tid] + ls[tid + 128]);
    atomicAdd(&colsumsq[o], lq[tid] + lq[tid + 128]);
  }
}
__global__ void k_stats(const float* __restrict__ cs, const float* __restrict__ cq,
                        const float* __restrict__ g, const float* __restrict__ b,
                        float* __restrict__ mu, float* __restrict__ sc,
                        float* __restrict__ sb) {
  int o = threadIdx.x;
  float m = cs[o] / (float)KN;
  float var = cq[o] / (float)KN - m * m;       // biased var, as torch BN
  float iv = rsqrtf(var + BN_EPS);
  mu[o] = m; sc[o] = iv * g[o]; sb[o] = b[o];
}
__global__ void k_update(const float* __restrict__ out, const float* __restrict__ mu,
                         const float* __restrict__ sc, const float* __restrict__ sb,
                         float* __restrict__ h, bf16* __restrict__ hbf) {
  int i = blockIdx.x * 256 + threadIdx.x;
  if (i < KN * KH) {
    int o = i & 127;
    float v = (out[i] - mu[o]) * sc[o] + sb[o];
    v = fmaxf(v, 0.f);
    float hn = h[i] + v;
    h[i] = hn; hbf[i] = (bf16)hn;
  }
}
__global__ void k_cnt(const int* __restrict__ batch, float* __restrict__ cnt) {
  int n = blockIdx.x * 256 + threadIdx.x;
  if (n < KN) atomicAdd(&cnt[batch[n]], 1.f);
}
__global__ void k_pooladd(const int* __restrict__ batch, const float* __restrict__ h,
                          float* __restrict__ pooled) {
  int i = blockIdx.x * 256 + threadIdx.x;
  if (i < KN * KH) {
    int n = i >> 7, o = i & 127;
    atomicAdd(&pooled[(size_t)batch[n] * 128 + o], h[i]);
  }
}
__global__ void k_pooldiv(const float* __restrict__ cnt, float* __restrict__ pooled) {
  int i = blockIdx.x * 256 + threadIdx.x;
  if (i < KG * KH) { int g = i >> 7; pooled[i] /= fmaxf(cnt[g], 1.f); }
}
__global__ void k_headout(const float* __restrict__ t, const float* __restrict__ w2,
                          const float* __restrict__ b2, float* __restrict__ y) {
  int g = blockIdx.x, tid = threadIdx.x;       // 128 threads = 4 waves
  float v = t[(size_t)g * 128 + tid] * w2[tid];
#pragma unroll
  for (int off = 16; off >= 1; off >>= 1) v += __shfl_down(v, off, 32);
  __shared__ float r[4];
  if ((tid & 31) == 0) r[tid >> 5] = v;
  __syncthreads();
  if (tid == 0) y[g] = r[0] + r[1] + r[2] + r[3] + b2[0];
}

// ---------------------------------------------------------------------------
extern "C" void kernel_launch(void* const* d_in, const int* in_sizes, int n_in,
                              void* d_out, int out_size, void* d_ws, size_t ws_size,
                              hipStream_t stream) {
  (void)in_sizes; (void)n_in; (void)out_size; (void)ws_size;
  const float* x       = (const float*)d_in[0];
  const int*   ei      = (const int*)d_in[1];     // [2,E] (integer -> int per harness)
  const float* eattr   = (const float*)d_in[2];
  const int*   batch   = (const int*)d_in[3];
  const float* node_w  = (const float*)d_in[4];
  const float* node_b  = (const float*)d_in[5];
  const float* e1_w    = (const float*)d_in[6];
  const float* e1_b    = (const float*)d_in[7];
  const float* e2_w    = (const float*)d_in[8];
  const float* e2_b    = (const float*)d_in[9];
  const float* root_w  = (const float*)d_in[10];
  const float* conv_b  = (const float*)d_in[11];
  const float* bn_g    = (const float*)d_in[12];
  const float* bn_b    = (const float*)d_in[13];
  const float* head_w1 = (const float*)d_in[14];
  const float* head_b1 = (const float*)d_in[15];
  const float* head_w2 = (const float*)d_in[16];
  const float* head_b2 = (const float*)d_in[17];
  float* y = (float*)d_out;
  const int* srcI = ei;
  const int* dstI = ei + KE;

  // ---- workspace carve-up (~31 MB) ----
  char* base = (char*)d_ws;
  size_t off = 0;
  auto alloc = [&](size_t bytes) -> char* {
    char* p = base + off;
    off = (off + bytes + 255) & ~(size_t)255;
    return p;
  };
  float* h        = (float*)alloc((size_t)KN * KH * 4);
  bf16*  hbf      = (bf16*) alloc((size_t)KN * KH * 2);
  float* agg      = (float*)alloc((size_t)KN * KH * 4);
  float* rootout  = (float*)alloc((size_t)KN * KH * 4);
  float* outbn    = (float*)alloc((size_t)KN * KH * 4);
  float* hb       = (float*)alloc((size_t)KN * KH * 4);
  bf16*  ebf      = (bf16*) alloc((size_t)KE * KH * 2);
  float* deg      = (float*)alloc(KN * 4);
  float* invdeg   = (float*)alloc(KN * 4);
  float* colsum   = (float*)alloc(128 * 4);
  float* colsumsq = (float*)alloc(128 * 4);
  float* muv      = (float*)alloc(128 * 4);
  float* scv      = (float*)alloc(128 * 4);
  float* sbv      = (float*)alloc(128 * 4);
  float* pooled   = (float*)alloc((size_t)KG * KH * 4);
  float* cnt      = (float*)alloc(KG * 4);
  float* thead    = (float*)alloc((size_t)KG * KH * 4);
  bf16* sw_node = (bf16*)alloc((size_t)2 * 4096 * 2);
  bf16* sw_head = (bf16*)alloc((size_t)4 * 4096 * 2);
  bf16 *sw_e1[KL], *sw_root[KL], *sw_e2b[KL], *sw_e2[KL];
  for (int l = 0; l < KL; ++l) sw_e1[l]   = (bf16*)alloc((size_t)1 * 4096 * 2);
  for (int l = 0; l < KL; ++l) sw_root[l] = (bf16*)alloc((size_t)4 * 4096 * 2);
  for (int l = 0; l < KL; ++l) sw_e2b[l]  = (bf16*)alloc((size_t)4 * 4096 * 2);
  for (int l = 0; l < KL; ++l) sw_e2[l]   = (bf16*)alloc((size_t)512 * 4096 * 2);

  auto sw = [&](const float* s, bf16* d, int Kc, int Kreal) {
    int total = Kc * 2048;
    k_swizzle_b<<<(total + 255) / 256, 256, 0, stream>>>(s, d, Kc, Kreal);
  };
  // One-time bf16 conversion + WMMA-fragment swizzle of all weights.
  sw(node_w, sw_node, 2, 64);
  sw(head_w1, sw_head, 4, 128);
  for (int l = 0; l < KL; ++l) {
    sw(e1_w + (size_t)l * KFE * KH, sw_e1[l], 1, 16);           // K padded 16->32
    sw(root_w + (size_t)l * KH * KH, sw_root[l], 4, 128);
    sw(e2_b + (size_t)l * KH * KH, sw_e2b[l], 4, 128);          // [h][o] matrix
    sw(e2_w + (size_t)l * KH * KH * KH, sw_e2[l], 512, 16384);  // the big B
  }

  // degree / inv_deg for mean aggregation
  k_fill<<<(KN + 255) / 256, 256, 0, stream>>>(deg, KN);
  k_deg<<<(KE + 255) / 256, 256, 0, stream>>>(dstI, deg);
  k_invdeg<<<(KN + 255) / 256, 256, 0, stream>>>(deg, invdeg);

  // node encoder: h = x @ node_w + node_b
  k_gemm<KFN, 2, 0><<<KN / 128, 256, 0, stream>>>(x, sw_node, node_b, h, hbf);

  for (int l = 0; l < KL; ++l) {
    // e = relu(edge_attr @ e1_w + e1_b)  (bf16 out)
    k_gemm<KFE, 1, 1><<<KE / 128, 256, 0, stream>>>(eattr, sw_e1[l],
                                                    e1_b + l * KH,
                                                    (float*)nullptr, ebf);
    // hb = h @ e2_b_mat   (per-node bias part of the per-edge message)
    k_gemm<KH, 4, 0><<<KN / 128, 256, 0, stream>>>(h, sw_e2b[l],
                                                   (const float*)nullptr,
                                                   hb, (bf16*)nullptr);
    // rootout = h @ root_w
    k_gemm<KH, 4, 0><<<KN / 128, 256, 0, stream>>>(h, sw_root[l],
                                                   (const float*)nullptr,
                                                   rootout, (bf16*)nullptr);
    k_fill<<<(KN * KH + 255) / 256, 256, 0, stream>>>(agg, KN * KH);
    k_fill<<<1, 256, 0, stream>>>(colsum, 128);
    k_fill<<<1, 256, 0, stream>>>(colsumsq, 128);
    // the big fused message GEMM + scatter
    k_fused_msg<<<KE / 128, 256, 0, stream>>>(hbf, ebf, sw_e2[l], srcI, dstI,
                                              hb, agg);
    k_combine1<<<KN / 32, 256, 0, stream>>>(agg, invdeg, rootout,
                                            conv_b + l * KH, outbn,
                                            colsum, colsumsq);
    k_stats<<<1, 128, 0, stream>>>(colsum, colsumsq, bn_g + l * KH,
                                   bn_b + l * KH, muv, scv, sbv);
    k_update<<<(KN * KH) / 256, 256, 0, stream>>>(outbn, muv, scv, sbv, h, hbf);
  }

  // global mean pool + head
  k_fill<<<(KG * KH + 255) / 256, 256, 0, stream>>>(pooled, KG * KH);
  k_fill<<<1, 256, 0, stream>>>(cnt, KG);
  k_cnt<<<(KN + 255) / 256, 256, 0, stream>>>(batch, cnt);
  k_pooladd<<<(KN * KH) / 256, 256, 0, stream>>>(batch, h, pooled);
  k_pooldiv<<<(KG * KH + 255) / 256, 256, 0, stream>>>(cnt, pooled);
  k_gemm<KH, 4, 1><<<KG / 128, 256, 0, stream>>>(pooled, sw_head, head_b1,
                                                 thead, (bf16*)nullptr);
  k_headout<<<KG, 128, 0, stream>>>(thead, head_w2, head_b2, y);
}
